// BitSelfAttention_43980465111749
// MI455X (gfx1250) — compile-verified, matrix operand order checked by reading, not run on hardware
//
#include <hip/hip_runtime.h>
#include <hip/hip_bf16.h>
#include <stdint.h>

// Problem constants (from reference)
#define T_DIM   2048
#define D_DIM   2048
#define DKV_DIM 512
#define HQ_N    32
#define HKV_N   8
#define HD      64   // head dim

typedef __attribute__((ext_vector_type(16))) __bf16 v16bf;
typedef __attribute__((ext_vector_type(8)))  float  v8f;

union FragU { v16bf v; uint4 q[2]; __bf16 h[16]; };

__device__ __forceinline__ __bf16 bf16_rn(float f) {
    uint32_t u = __builtin_bit_cast(uint32_t, f);
    u += 0x7FFFu + ((u >> 16) & 1u);          // round-to-nearest-even
    unsigned short hv = (unsigned short)(u >> 16);
    return __builtin_bit_cast(__bf16, hv);
}

__device__ __forceinline__ unsigned short bf16_bits(float f) {
    uint32_t u = __builtin_bit_cast(uint32_t, f);
    u += 0x7FFFu + ((u >> 16) & 1u);
    return (unsigned short)(u >> 16);
}

// ----------------------------------------------------------------- DPP helpers
// Butterfly reduction within each 16-lane half (a DPP "row"), pure VALU:
//   quad_perm(1,0,3,2) -> quad_perm(2,3,0,1) -> row_half_mirror -> row_mirror
// After steps 1/2 values are uniform over 2/4 lanes, so the mirrors act as
// xor4/xor8. No LDS, no dscnt waits (vs ds_bpermute from __shfl_xor).
template<int CTRL>
__device__ __forceinline__ float dpp_movf(float x) {
    int r = __builtin_amdgcn_update_dpp(0, __builtin_bit_cast(int, x),
                                        CTRL, 0xF, 0xF, true);
    return __builtin_bit_cast(float, r);
}
__device__ __forceinline__ float row16_max(float x) {
    x = fmaxf(x, dpp_movf<0xB1>(x));   // quad_perm [1,0,3,2]
    x = fmaxf(x, dpp_movf<0x4E>(x));   // quad_perm [2,3,0,1]
    x = fmaxf(x, dpp_movf<0x141>(x));  // row_half_mirror
    x = fmaxf(x, dpp_movf<0x140>(x));  // row_mirror
    return x;
}
__device__ __forceinline__ float row16_sum(float x) {
    x += dpp_movf<0xB1>(x);
    x += dpp_movf<0x4E>(x);
    x += dpp_movf<0x141>(x);
    x += dpp_movf<0x140>(x);
    return x;
}

// Load a 16x32 bf16 operand fragment (A layout, or B as N-major x K) from a
// row-major [16+][ld] tile. Per CDNA5 ISA 7.12.2:
//   lanes 0-15 : row = lane,    halves 0-7 = K 0..7,   halves 8-15 = K 16..23
//   lanes 16-31: row = lane-16, halves 0-7 = K 8..15,  halves 8-15 = K 24..31
__device__ __forceinline__ v16bf load_frag(const __bf16* base, int ld) {
    const int lane = threadIdx.x & 31;
    const __bf16* p = base + (lane & 15) * ld + ((lane >> 4) << 3);
    FragU f;
    f.q[0] = *reinterpret_cast<const uint4*>(p);
    f.q[1] = *reinterpret_cast<const uint4*>(p + 16);
    return f.v;
}

__device__ __forceinline__ v8f wmma_bf16(v16bf a, v16bf b, v8f c) {
    return __builtin_amdgcn_wmma_f32_16x16x32_bf16(
        /*neg_a=*/false, a, /*neg_b=*/false, b,
        /*c_mod=*/(short)0, c, /*reuse_a=*/false, /*reuse_b=*/false);
}

// ---------------------------------------------------------------- converters
// 4 elements / thread: float4 in, two packed bf16 dwords out.
__global__ void cvt_f32_bf16_v4(const float* __restrict__ in,
                                __bf16* __restrict__ out, size_t n4) {
    size_t i = (size_t)blockIdx.x * blockDim.x + threadIdx.x;
    if (i < n4) {
        float4 v = reinterpret_cast<const float4*>(in)[i];
        uint2 p;
        p.x = (uint32_t)bf16_bits(v.x) | ((uint32_t)bf16_bits(v.y) << 16);
        p.y = (uint32_t)bf16_bits(v.z) | ((uint32_t)bf16_bits(v.w) << 16);
        reinterpret_cast<uint2*>(out)[i] = p;
    }
}

// out[c*rows + r] = in[r*cols + c]  (build W^T in N-major x K layout)
__global__ void cvt_f32_bf16_T(const float* __restrict__ in,
                               __bf16* __restrict__ out, int rows, int cols) {
    size_t i = (size_t)blockIdx.x * blockDim.x + threadIdx.x;
    if (i < (size_t)rows * cols) {
        int r = (int)(i / cols), c = (int)(i % cols);
        out[(size_t)c * rows + r] = bf16_rn(in[i]);
    }
}

// ------------------------------------------------------------------- GEMM
// C[M,N] = A[M,K] (row major) * B[N,K]^T (B stored N-major x K), fp32 accum.
// Block = 256 threads = 8 waves; tile 128x128, K-step 32, double-buffered LDS:
// next tile's global loads issue before the current tile's 8 WMMAs, so the
// loadcnt wait is covered by matrix work. One barrier per K-step.
template<bool F32OUT>
__global__ __launch_bounds__(256)
void gemm_bf16_wmma(const __bf16* __restrict__ A, const __bf16* __restrict__ B,
                    __bf16* __restrict__ Cb, float* __restrict__ Cf,
                    int M, int N, int K) {
    __shared__ __bf16 As[2][128 * 32];
    __shared__ __bf16 Bs[2][128 * 32];

    const int tid  = threadIdx.x;
    const int lane = tid & 31;
    const int wid  = tid >> 5;
    const int m0   = blockIdx.y * 128;
    const int n0   = blockIdx.x * 128;
    const int wm   = (wid >> 2) * 64;
    const int wn   = (wid & 3) * 32;

    // staging coordinates: 2 x 16B per matrix per thread
    const int r0 = tid >> 2,          c0 = (tid & 3) << 3;
    const int r1 = (tid + 256) >> 2,  c1 = ((tid + 256) & 3) << 3;

    v8f acc[4][2];
#pragma unroll
    for (int i = 0; i < 4; ++i)
#pragma unroll
        for (int j = 0; j < 2; ++j)
#pragma unroll
            for (int e = 0; e < 8; ++e) acc[i][j][e] = 0.0f;

    const int nk = K >> 5;
    uint4 ra0, ra1, rb0, rb1;

    // prologue: fetch tile 0 and stage into buffer 0
    ra0 = *reinterpret_cast<const uint4*>(&A[(size_t)(m0 + r0) * K + c0]);
    ra1 = *reinterpret_cast<const uint4*>(&A[(size_t)(m0 + r1) * K + c1]);
    rb0 = *reinterpret_cast<const uint4*>(&B[(size_t)(n0 + r0) * K + c0]);
    rb1 = *reinterpret_cast<const uint4*>(&B[(size_t)(n0 + r1) * K + c1]);
    *reinterpret_cast<uint4*>(&As[0][r0 * 32 + c0]) = ra0;
    *reinterpret_cast<uint4*>(&As[0][r1 * 32 + c1]) = ra1;
    *reinterpret_cast<uint4*>(&Bs[0][r0 * 32 + c0]) = rb0;
    *reinterpret_cast<uint4*>(&Bs[0][r1 * 32 + c1]) = rb1;
    __syncthreads();

    for (int kt = 0; kt < nk; ++kt) {
        const int cur = kt & 1;
        if (kt + 1 < nk) {                   // prefetch next tile into regs
            const int k0 = (kt + 1) << 5;
            ra0 = *reinterpret_cast<const uint4*>(&A[(size_t)(m0 + r0) * K + k0 + c0]);
            ra1 = *reinterpret_cast<const uint4*>(&A[(size_t)(m0 + r1) * K + k0 + c1]);
            rb0 = *reinterpret_cast<const uint4*>(&B[(size_t)(n0 + r0) * K + k0 + c0]);
            rb1 = *reinterpret_cast<const uint4*>(&B[(size_t)(n0 + r1) * K + k0 + c1]);
        }

        v16bf af[4], bf2[2];
#pragma unroll
        for (int i = 0; i < 4; ++i) af[i]  = load_frag(&As[cur][(wm + i * 16) * 32], 32);
#pragma unroll
        for (int j = 0; j < 2; ++j) bf2[j] = load_frag(&Bs[cur][(wn + j * 16) * 32], 32);
#pragma unroll
        for (int i = 0; i < 4; ++i)
#pragma unroll
            for (int j = 0; j < 2; ++j)
                acc[i][j] = wmma_bf16(af[i], bf2[j], acc[i][j]);

        if (kt + 1 < nk) {
            const int nxt = cur ^ 1;
            *reinterpret_cast<uint4*>(&As[nxt][r0 * 32 + c0]) = ra0;
            *reinterpret_cast<uint4*>(&As[nxt][r1 * 32 + c1]) = ra1;
            *reinterpret_cast<uint4*>(&Bs[nxt][r0 * 32 + c0]) = rb0;
            *reinterpret_cast<uint4*>(&Bs[nxt][r1 * 32 + c1]) = rb1;
            __syncthreads();
        }
    }

    // Store. C layout: VGPR e -> row e (lanes 0-15) / row e+8 (lanes 16-31),
    // col = lane & 15.
    const int rowoff = (lane >> 4) << 3;
#pragma unroll
    for (int i = 0; i < 4; ++i) {
        int rbase = m0 + wm + i * 16 + rowoff;
#pragma unroll
        for (int j = 0; j < 2; ++j) {
            int n = n0 + wn + j * 16 + (lane & 15);
#pragma unroll
            for (int e = 0; e < 8; ++e) {
                size_t off = (size_t)(rbase + e) * N + n;
                if constexpr (F32OUT) Cf[off] = acc[i][j][e];
                else                  Cb[off] = bf16_rn(acc[i][j][e]);
            }
        }
    }
}

// ------------------------------------------------------------ flash attention
// grid = (T/64, HQ); block = 128 threads (4 waves). Wave w handles 16 q-rows.
// Online (flash) softmax over 32-key steps, causal. Row reductions are DPP.
__global__ __launch_bounds__(128)
void attn_fwd(const __bf16* __restrict__ Q, const __bf16* __restrict__ Km,
              const __bf16* __restrict__ Vm, __bf16* __restrict__ O) {
    __shared__ __bf16 Ks[32 * 64];      // [key][d]
    __shared__ __bf16 Vt[64 * 32];      // [d][key]  (transposed V)
    __shared__ __bf16 Ps[4 * 16 * 32];  // per-wave P tile [row][key]

    const int tid   = threadIdx.x;
    const int lane  = tid & 31;
    const int wid   = tid >> 5;
    const int qt    = blockIdx.x;
    const int h     = blockIdx.y;
    const int kvh   = h >> 2;                 // group = HQ/HKV = 4
    const int q0    = qt * 64;
    const int qrow0 = q0 + wid * 16;
    const int rowoff = (lane >> 4) << 3;      // 0 or 8
    const int nidx   = lane & 15;

    // Q fragments for this wave's 16 rows, head dims [0..63] -> two K=32 frags
    const __bf16* qbase = Q + (size_t)qrow0 * D_DIM + h * HD;
    v16bf qf0 = load_frag(qbase, D_DIM);
    v16bf qf1 = load_frag(qbase + 32, D_DIM);

    float Mrow[8], Lrow[8];
    v8f o[4];
#pragma unroll
    for (int e = 0; e < 8; ++e) { Mrow[e] = -1e30f; Lrow[e] = 0.0f; }
#pragma unroll
    for (int j = 0; j < 4; ++j)
#pragma unroll
        for (int e = 0; e < 8; ++e) o[j][e] = 0.0f;

    const int nkb = (q0 + 64) / 32;           // causal: keys up to last q row
    for (int kb = 0; kb < nkb; ++kb) {
        const int kg0 = kb * 32;

        // stage K tile [32 keys][64 dims]
#pragma unroll
        for (int s = 0; s < 2; ++s) {
            int idx = tid + s * 128;          // 0..255
            int r = idx >> 3, c = (idx & 7) << 3;
            *reinterpret_cast<uint4*>(&Ks[r * 64 + c]) =
                *reinterpret_cast<const uint4*>(
                    &Km[(size_t)(kg0 + r) * DKV_DIM + kvh * HD + c]);
        }
        // stage V transposed: Vt[d][key]
        {
            int key = tid >> 2;               // 0..31
            int d0  = (tid & 3) << 4;         // 0,16,32,48
            FragU f;
            const __bf16* vsrc = &Vm[(size_t)(kg0 + key) * DKV_DIM + kvh * HD + d0];
            f.q[0] = *reinterpret_cast<const uint4*>(vsrc);
            f.q[1] = *reinterpret_cast<const uint4*>(vsrc + 8);
#pragma unroll
            for (int j = 0; j < 16; ++j) Vt[(d0 + j) * 32 + key] = f.h[j];
        }
        __syncthreads();

        // S = Q K^T : 16 rows x 32 keys (two 16x16 accs), K-dim = 64 -> 2 wmma each
        v8f s0, s1;
#pragma unroll
        for (int e = 0; e < 8; ++e) { s0[e] = 0.0f; s1[e] = 0.0f; }
        s0 = wmma_bf16(qf0, load_frag(&Ks[0], 64), s0);
        s0 = wmma_bf16(qf1, load_frag(&Ks[32], 64), s0);
        s1 = wmma_bf16(qf0, load_frag(&Ks[16 * 64], 64), s1);
        s1 = wmma_bf16(qf1, load_frag(&Ks[16 * 64 + 32], 64), s1);

        // online softmax in C layout; each row lives in one 16-lane half
        __bf16* pw = &Ps[wid * 16 * 32];
#pragma unroll
        for (int e = 0; e < 8; ++e) {
            int qg = qrow0 + rowoff + e;
            float x0 = s0[e] * 0.125f;        // 1/sqrt(64)
            float x1 = s1[e] * 0.125f;
            if (kg0 + nidx      > qg) x0 = -1e30f;
            if (kg0 + 16 + nidx > qg) x1 = -1e30f;
            float rm   = row16_max(fmaxf(x0, x1));
            float mnew = fmaxf(Mrow[e], rm);
            float corr = __expf(Mrow[e] - mnew);
            float p0 = __expf(x0 - mnew);
            float p1 = __expf(x1 - mnew);
            float rs = row16_sum(p0 + p1);
            Lrow[e] = Lrow[e] * corr + rs;
            Mrow[e] = mnew;
#pragma unroll
            for (int j = 0; j < 4; ++j) o[j][e] *= corr;
            int prow = e + rowoff;
            pw[prow * 32 + nidx]      = bf16_rn(p0);
            pw[prow * 32 + 16 + nidx] = bf16_rn(p1);
        }

        // O += P(16x32) * V(32x64): A frag from LDS P, B frags from Vt (N=d, K=key)
        v16bf pf = load_frag(pw, 32);
#pragma unroll
        for (int j = 0; j < 4; ++j)
            o[j] = wmma_bf16(pf, load_frag(&Vt[j * 16 * 32], 32), o[j]);
        __syncthreads();
    }

    // normalize and write ctx[row, h*64 + d] (bf16)
#pragma unroll
    for (int j = 0; j < 4; ++j) {
        int d = h * HD + j * 16 + nidx;
#pragma unroll
        for (int e = 0; e < 8; ++e) {
            int row = qrow0 + rowoff + e;
            O[(size_t)row * D_DIM + d] = bf16_rn(o[j][e] / Lrow[e]);
        }
    }
}

// ------------------------------------------------------------------ launcher
extern "C" void kernel_launch(void* const* d_in, const int* in_sizes, int n_in,
                              void* d_out, int out_size, void* d_ws, size_t ws_size,
                              hipStream_t stream) {
    (void)in_sizes; (void)n_in; (void)out_size; (void)ws_size;
    const float* x  = (const float*)d_in[0];
    const float* wq = (const float*)d_in[1];
    const float* wk = (const float*)d_in[2];
    const float* wv = (const float*)d_in[3];
    const float* wo = (const float*)d_in[4];
    float* out = (float*)d_out;

    const size_t TD = (size_t)T_DIM * D_DIM;      // 4M
    const size_t DD = (size_t)D_DIM * D_DIM;      // 4M
    const size_t KD = (size_t)DKV_DIM * D_DIM;    // 1M
    const size_t TK = (size_t)T_DIM * DKV_DIM;    // 1M

    __bf16* xb  = (__bf16*)d_ws;      // also reused as ctx after attention
    __bf16* wqb = xb  + TD;
    __bf16* wkb = wqb + DD;
    __bf16* wvb = wkb + KD;
    __bf16* wob = wvb + KD;
    __bf16* Qb  = wob + DD;
    __bf16* Kb  = Qb  + TD;
    __bf16* Vb  = Kb  + TK;
    __bf16* ctx = xb;                  // x dead after V projection

    const int thr = 256;
    cvt_f32_bf16_v4<<<dim3((unsigned)((TD / 4 + thr - 1) / thr)), thr, 0, stream>>>(x,  xb,  TD / 4);
    cvt_f32_bf16_v4<<<dim3((unsigned)((DD / 4 + thr - 1) / thr)), thr, 0, stream>>>(wq, wqb, DD / 4);
    cvt_f32_bf16_v4<<<dim3((unsigned)((KD / 4 + thr - 1) / thr)), thr, 0, stream>>>(wk, wkb, KD / 4);
    cvt_f32_bf16_v4<<<dim3((unsigned)((KD / 4 + thr - 1) / thr)), thr, 0, stream>>>(wv, wvb, KD / 4);
    cvt_f32_bf16_T<<<dim3((unsigned)((DD + thr - 1) / thr)), thr, 0, stream>>>(wo, wob, D_DIM, D_DIM);

    // Q = x Wq^T  [2048 x 2048]
    gemm_bf16_wmma<false><<<dim3(D_DIM / 128, T_DIM / 128), 256, 0, stream>>>(
        xb, wqb, Qb, nullptr, T_DIM, D_DIM, D_DIM);
    // K = x Wk^T  [2048 x 512]
    gemm_bf16_wmma<false><<<dim3(DKV_DIM / 128, T_DIM / 128), 256, 0, stream>>>(
        xb, wkb, Kb, nullptr, T_DIM, DKV_DIM, D_DIM);
    // V = x Wv^T  [2048 x 512]
    gemm_bf16_wmma<false><<<dim3(DKV_DIM / 128, T_DIM / 128), 256, 0, stream>>>(
        xb, wvb, Vb, nullptr, T_DIM, DKV_DIM, D_DIM);

    // causal GQA flash attention -> ctx (reuses xb)
    attn_fwd<<<dim3(T_DIM / 64, HQ_N), 128, 0, stream>>>(Qb, Kb, Vb, ctx);

    // out = ctx @ Wo  (Wo pre-transposed to N-major x K), fp32 output
    gemm_bf16_wmma<true><<<dim3(D_DIM / 128, T_DIM / 128), 256, 0, stream>>>(
        ctx, wob, nullptr, out, T_DIM, D_DIM, D_DIM);
}
